// TopKGate_38044820308413
// MI455X (gfx1250) — matrix operand order repaired and checked by reading, not compiled, not run on hardware
//
#include <hip/hip_runtime.h>
#include <stdint.h>

// ---------------- problem constants ----------------
#define S_TOK 16384
#define MDIM  1024
#define GEXP  64
#define KTOP  2
#define CAPAC 512   // K * ceil(S/GE)

// ---------------- WMMA types ----------------
typedef __bf16 bf16_t;
typedef bf16_t v16bf __attribute__((ext_vector_type(16)));
typedef float  v8f   __attribute__((ext_vector_type(8)));

__device__ __forceinline__ unsigned short f2bf(float f) {
    union { float f; uint32_t u; } v; v.f = f;
    uint32_t u = v.u;
    uint32_t r = u + 0x7FFFu + ((u >> 16) & 1u);   // round-to-nearest-even
    return (unsigned short)(r >> 16);
}

// ---- CDNA5 async global->LDS copy (ASYNCcnt-tracked, no VGPR round trip) ----
__device__ __forceinline__ void async_load_b128(unsigned lds_off, const void* gaddr) {
    asm volatile("global_load_async_to_lds_b128 %0, %1, off"
                 :: "v"(lds_off), "v"(gaddr) : "memory");
}
__device__ __forceinline__ void wait_async0() {
    asm volatile("s_wait_asynccnt 0" ::: "memory");
}
// low 32 bits of a shared-aperture generic address == LDS byte offset
__device__ __forceinline__ unsigned lds_off32(const void* p) {
    return (unsigned)(uintptr_t)p;
}

// ============================================================
// 1) Gating: logits, softmax, top-2, renormalized top-k gates
// ============================================================
__global__ __launch_bounds__(64) void gate_kernel(
    const float* __restrict__ x, const float* __restrict__ wg,
    float* __restrict__ gates, int* __restrict__ topk, float* __restrict__ gk)
{
    __shared__ float xs[MDIM];
    __shared__ float lg[GEXP];
    __shared__ float red[2];
    const int s = blockIdx.x, t = threadIdx.x;
    const float* xr = x + (size_t)s * MDIM;
    for (int i = t; i < MDIM / 4; i += 64)
        ((float4*)xs)[i] = ((const float4*)xr)[i];
    __syncthreads();

    const float* w = wg + (size_t)t * MDIM;
    float acc = 0.f;
    for (int m = 0; m < MDIM; m += 4)
        acc += xs[m]*w[m] + xs[m+1]*w[m+1] + xs[m+2]*w[m+2] + xs[m+3]*w[m+3];
    lg[t] = acc;
    __syncthreads();

    if (t == 0) {
        float mx = lg[0];
        for (int e = 1; e < GEXP; ++e) mx = fmaxf(mx, lg[e]);
        float sum = 0.f;
        for (int e = 0; e < GEXP; ++e) sum += expf(lg[e] - mx);
        red[0] = mx; red[1] = sum;
    }
    __syncthreads();

    const float g = expf(lg[t] - red[0]) / red[1];
    gates[(size_t)s * GEXP + t] = g;

    if (t == 0) {
        float v0 = -INFINITY, v1 = -INFINITY; int i0 = 0, i1 = 0;
        for (int e = 0; e < GEXP; ++e) {
            float v = lg[e];
            if (v > v0)      { v1 = v0; i1 = i0; v0 = v; i0 = e; }
            else if (v > v1) { v1 = v;  i1 = e; }
        }
        float g0 = expf(v0 - red[0]) / red[1];
        float g1 = expf(v1 - red[0]) / red[1];
        float den = fmaxf(g0 + g1, 1.1920929e-7f);   // FLT_EPSILON clip
        topk[s*2]   = i0; topk[s*2+1] = i1;
        gk[s*2]     = g0 / den;
        gk[s*2+1]   = g1 / den;
    }
}

// ============================================================
// 2) me[e] = sum_s gates[s][e]  (deterministic reduction)
// ============================================================
__global__ __launch_bounds__(256) void me_kernel(
    const float* __restrict__ gates, float* __restrict__ me)
{
    __shared__ float r[256];
    const int e = blockIdx.x, t = threadIdx.x;
    float a = 0.f;
    for (int s = t; s < S_TOK; s += 256) a += gates[(size_t)s * GEXP + e];
    r[t] = a; __syncthreads();
    for (int w = 128; w > 0; w >>= 1) { if (t < w) r[t] += r[t + w]; __syncthreads(); }
    if (t == 0) me[e] = r[0];
}

// ============================================================
// 3) Capacity locations: exclusive cumsum per (k, expert),
//    k=1 offset by k=0 totals. One wave per expert, ballot scan.
// ============================================================
__global__ __launch_bounds__(32) void loc_kernel(
    const int* __restrict__ topk, int* __restrict__ loc, float* __restrict__ ce)
{
    const int e = blockIdx.x;
    const int lane = threadIdx.x;
    const unsigned int below = (1u << lane) - 1u;
    int cnt = 0;
    for (int k = 0; k < KTOP; ++k) {
        if (k == 1 && lane == 0) ce[e] = (float)cnt;   // top-1 counts (ce)
        for (int s0 = 0; s0 < S_TOK; s0 += 32) {
            const int s = s0 + lane;
            const bool p = (topk[s * 2 + k] == e);
            const unsigned int b = (unsigned int)__ballot(p);
            if (p) loc[s * 2 + k] = cnt + __popc(b & below);
            cnt += __popc(b);
        }
    }
}

// ============================================================
// 4) Zero dispatch buffer + scatter tokens + we fp32->bf16
// ============================================================
__global__ void zero_kernel(uint4* __restrict__ p, long n)
{
    long i = (long)blockIdx.x * blockDim.x + threadIdx.x;
    const long st = (long)gridDim.x * blockDim.x;
    const uint4 z = {0u, 0u, 0u, 0u};
    for (; i < n; i += st) p[i] = z;
}

__global__ __launch_bounds__(256) void convert_we_kernel(
    const float* __restrict__ we, unsigned short* __restrict__ web, long n8)
{
    long i = (long)blockIdx.x * blockDim.x + threadIdx.x;
    const long st = (long)gridDim.x * blockDim.x;
    for (; i < n8; i += st) {
        float4 a = ((const float4*)we)[i * 2];
        float4 b = ((const float4*)we)[i * 2 + 1];
        union { unsigned short h[8]; uint4 v; } pk;
        pk.h[0]=f2bf(a.x); pk.h[1]=f2bf(a.y); pk.h[2]=f2bf(a.z); pk.h[3]=f2bf(a.w);
        pk.h[4]=f2bf(b.x); pk.h[5]=f2bf(b.y); pk.h[6]=f2bf(b.z); pk.h[7]=f2bf(b.w);
        ((uint4*)web)[i] = pk.v;
    }
}

__global__ __launch_bounds__(128) void dispatch_kernel(
    const float* __restrict__ x, const int* __restrict__ topk,
    const int* __restrict__ loc, unsigned short* __restrict__ disp)
{
    const int sk = blockIdx.x;
    const int s = sk >> 1, k = sk & 1;
    const int l = loc[s * 2 + k];
    if (l >= CAPAC) return;                        // overflow -> dropped
    const int e = topk[s * 2 + k];
    const float* src = x + (size_t)s * MDIM;
    unsigned short* dst = disp + ((size_t)e * CAPAC + l) * MDIM;
    for (int i = threadIdx.x * 8; i < MDIM; i += 128 * 8) {
        float4 a = *(const float4*)(src + i);
        float4 b = *(const float4*)(src + i + 4);
        union { unsigned short h[8]; uint4 v; } pk;
        pk.h[0]=f2bf(a.x); pk.h[1]=f2bf(a.y); pk.h[2]=f2bf(a.z); pk.h[3]=f2bf(a.w);
        pk.h[4]=f2bf(b.x); pk.h[5]=f2bf(b.y); pk.h[6]=f2bf(b.z); pk.h[7]=f2bf(b.w);
        *(uint4*)(dst + i) = pk.v;
    }
}

// ============================================================
// 5) Grouped expert GEMM:  out[g] = disp[g] (bf16) @ web[g] (bf16)
//    BM=128 BN=128 BK=32, 8 waves/block, wave = 32x64 sub-tile.
//    Double-buffered LDS filled by global_load_async_to_lds_b128,
//    v_wmma_f32_16x16x32_bf16 with fp32 accumulate.
// ============================================================
#define BM 128
#define BN 128
#define BK 32
#define LDA (BK + 8)   // ushort stride, 80B rows (16B aligned chunks)
#define LDB (BN + 8)   // ushort stride, 272B rows (16B aligned chunks)

__global__ __launch_bounds__(256) void expert_gemm_kernel(
    const unsigned short* __restrict__ disp, const unsigned short* __restrict__ web,
    float* __restrict__ out)
{
    __shared__ __align__(16) unsigned short As[2][BM * LDA];
    __shared__ __align__(16) unsigned short Bs[2][BK * LDB];

    const int bx = blockIdx.x;
    const int g   = bx >> 5;          // 32 tiles per expert (4 x 8)
    const int rem = bx & 31;
    const int tm  = rem >> 3;         // CAP/BM   = 4
    const int tn  = rem & 7;          // MDIM/BN  = 8

    const unsigned short* A0 = disp + ((size_t)g * CAPAC + (size_t)tm * BM) * MDIM;
    const unsigned short* B0 = web  + (size_t)g * MDIM * MDIM + (size_t)tn * BN;

    const int tid  = threadIdx.x;
    const int lane = tid & 31;
    const int wv   = tid >> 5;
    const int wm   = wv & 3;          // wave row slot  (32 rows)
    const int wn   = wv >> 2;         // wave col slot  (64 cols)

    // ---- per-thread async-copy chunk assignment (16B chunks) ----
    // A tile: 128 rows x 64B = 512 chunks; B tile: 32 rows x 256B = 512 chunks
    const int ia0 = tid, ia1 = tid + 256;
    const int rA0 = ia0 >> 2, cA0 = (ia0 & 3) * 8;
    const int rA1 = ia1 >> 2, cA1 = (ia1 & 3) * 8;
    const int rB0 = ia0 >> 4, cB0 = (ia0 & 15) * 8;
    const int rB1 = ia1 >> 4, cB1 = (ia1 & 15) * 8;

    const unsigned short* gA0 = A0 + (size_t)rA0 * MDIM + cA0;
    const unsigned short* gA1 = A0 + (size_t)rA1 * MDIM + cA1;
    const unsigned short* gB0 = B0 + (size_t)rB0 * MDIM + cB0;
    const unsigned short* gB1 = B0 + (size_t)rB1 * MDIM + cB1;

    const unsigned lA0 = lds_off32(&As[0][rA0 * LDA + cA0]);
    const unsigned lA1 = lds_off32(&As[0][rA1 * LDA + cA1]);
    const unsigned lB0 = lds_off32(&Bs[0][rB0 * LDB + cB0]);
    const unsigned lB1 = lds_off32(&Bs[0][rB1 * LDB + cB1]);
    const unsigned ABUF = (unsigned)(BM * LDA * sizeof(unsigned short));
    const unsigned BBUF = (unsigned)(BK * LDB * sizeof(unsigned short));

    const v8f vzero = {0.f,0.f,0.f,0.f,0.f,0.f,0.f,0.f};
    v8f acc[2][4];
    #pragma unroll
    for (int i = 0; i < 2; ++i)
        #pragma unroll
        for (int j = 0; j < 4; ++j) acc[i][j] = vzero;

    // ---- prologue: fill buffer 0 for k0 = 0 ----
    async_load_b128(lA0, gA0);
    async_load_b128(lA1, gA1);
    async_load_b128(lB0, gB0);
    async_load_b128(lB1, gB1);
    wait_async0();
    __syncthreads();

    int buf = 0;
    for (int k0 = 0; k0 < MDIM; k0 += BK) {
        // ---- issue async copies for next K-slab into the other buffer ----
        if (k0 + BK < MDIM) {
            const int kn = k0 + BK;                 // A advances K columns
            const unsigned ob = (unsigned)(buf ^ 1);
            async_load_b128(lA0 + ob * ABUF, gA0 + kn);
            async_load_b128(lA1 + ob * ABUF, gA1 + kn);
            async_load_b128(lB0 + ob * BBUF, gB0 + (size_t)kn * MDIM);  // B advances K rows
            async_load_b128(lB1 + ob * BBUF, gB1 + (size_t)kn * MDIM);
        }

        // ---- fragments per documented CDNA5 16-bit layouts ----
        // A (16x32): lanes 0-15 -> M=0..15, K in {0..7,16..23}; lanes 16-31 -> K in {8..15,24..31}
        // B (32x16): lane l -> N=l%16, K in {0..7,16..23} (lanes 0-15) / {8..15,24..31} (lanes 16-31)
        v16bf af[2], bfr[4];
        {
            const int r  = lane & 15;
            const int kb = (lane < 16) ? 0 : 8;
            #pragma unroll
            for (int mi = 0; mi < 2; ++mi) {
                const unsigned short* p = &As[buf][(wm * 32 + mi * 16 + r) * LDA + kb];
                union { uint4 q[2]; v16bf v; } u;
                u.q[0] = *(const uint4*)p;
                u.q[1] = *(const uint4*)(p + 16);
                af[mi] = u.v;
            }
            #pragma unroll
            for (int ni = 0; ni < 4; ++ni) {
                const unsigned short* p = &Bs[buf][kb * LDB + wn * 64 + ni * 16 + r];
                union { unsigned short h[16]; v16bf v; } u;
                #pragma unroll
                for (int s = 0; s < 8; ++s) {
                    u.h[s]     = p[s * LDB];
                    u.h[s + 8] = p[(s + 16) * LDB];
                }
                bfr[ni] = u.v;
            }
        }
        #pragma unroll
        for (int mi = 0; mi < 2; ++mi)
            #pragma unroll
            for (int ni = 0; ni < 4; ++ni)
                acc[mi][ni] = __builtin_amdgcn_wmma_f32_16x16x32_bf16(
                    false, af[mi], false, bfr[ni],
                    (short)0, acc[mi][ni], false, false);

        // ---- my async copies landed; everyone done reading buf & filling buf^1 ----
        wait_async0();
        __syncthreads();
        buf ^= 1;
    }

    // ---- epilogue: C/D layout — VGPR i: M = i (lanes 0-15) / 8+i (lanes 16-31), N = lane%16
    const int r0 = (lane < 16) ? 0 : 8;
    const int cc = lane & 15;
    #pragma unroll
    for (int mi = 0; mi < 2; ++mi)
        #pragma unroll
        for (int ni = 0; ni < 4; ++ni) {
            const int row = tm * BM + wm * 32 + mi * 16 + r0;
            const int col = tn * BN + wn * 64 + ni * 16 + cc;
            float* op = out + ((size_t)g * CAPAC + row) * MDIM + col;
            #pragma unroll
            for (int i = 0; i < 8; ++i) op[(size_t)i * MDIM] = acc[mi][ni][i];
        }
}

// ============================================================
// 6) Combine: y[s] = w0 * eo[slot0] + w1 * eo[slot1]
// ============================================================
__global__ __launch_bounds__(128) void combine_kernel(
    const float* __restrict__ eo, const int* __restrict__ topk,
    const int* __restrict__ loc, const float* __restrict__ gk,
    float* __restrict__ y)
{
    const int s = blockIdx.x, t = threadIdx.x;
    const int e0 = topk[s*2],   e1 = topk[s*2+1];
    const int l0 = loc[s*2],    l1 = loc[s*2+1];
    const float w0 = (l0 < CAPAC) ? gk[s*2]   : 0.f;
    const float w1 = (l1 < CAPAC) ? gk[s*2+1] : 0.f;
    const float* p0 = (l0 < CAPAC) ? eo + ((size_t)e0 * CAPAC + l0) * MDIM : eo;
    const float* p1 = (l1 < CAPAC) ? eo + ((size_t)e1 * CAPAC + l1) * MDIM : eo;
    float* yr = y + (size_t)s * MDIM;
    for (int i = t * 4; i < MDIM; i += 128 * 4) {
        float4 a = *(const float4*)(p0 + i);
        float4 b = *(const float4*)(p1 + i);
        float4 r;
        r.x = w0*a.x + w1*b.x; r.y = w0*a.y + w1*b.y;
        r.z = w0*a.z + w1*b.z; r.w = w0*a.w + w1*b.w;
        *(float4*)(yr + i) = r;
    }
}

// ============================================================
// 7) Aux loss: sum(me * ce) * GE / S^2
// ============================================================
__global__ __launch_bounds__(64) void loss_kernel(
    const float* __restrict__ me, const float* __restrict__ ce,
    float* __restrict__ out)
{
    __shared__ float r[GEXP];
    const int t = threadIdx.x;
    r[t] = me[t] * ce[t];
    __syncthreads();
    for (int w = 32; w > 0; w >>= 1) { if (t < w) r[t] += r[t + w]; __syncthreads(); }
    if (t == 0) out[0] = r[0] * ((float)GEXP / ((float)S_TOK * (float)S_TOK));
}

// ============================================================
// launch
// ============================================================
extern "C" void kernel_launch(void* const* d_in, const int* in_sizes, int n_in,
                              void* d_out, int out_size, void* d_ws, size_t ws_size,
                              hipStream_t stream)
{
    const float* x  = (const float*)d_in[0];   // [S, M]
    const float* wg = (const float*)d_in[1];   // [GE, M]
    const float* we = (const float*)d_in[2];   // [GE, M, M]
    float* y     = (float*)d_out;              // [S, M]
    float* lloss = y + (size_t)S_TOK * MDIM;   // scalar (last element)

    char* ws = (char*)d_ws;
    size_t o = 0;
    auto alloc = [&](size_t bytes) -> char* {
        char* p = ws + o;
        o = (o + bytes + 255) & ~(size_t)255;
        return p;
    };
    float* gates        = (float*)alloc((size_t)S_TOK * GEXP * 4);
    int*   topk         = (int*)  alloc((size_t)S_TOK * KTOP * 4);
    float* gk           = (float*)alloc((size_t)S_TOK * KTOP * 4);
    int*   loc          = (int*)  alloc((size_t)S_TOK * KTOP * 4);
    float* me           = (float*)alloc(GEXP * 4);
    float* ce           = (float*)alloc(GEXP * 4);
    unsigned short* disp= (unsigned short*)alloc((size_t)GEXP * CAPAC * MDIM * 2);
    unsigned short* web = (unsigned short*)alloc((size_t)GEXP * MDIM * MDIM * 2);
    float* eo           = (float*)alloc((size_t)GEXP * CAPAC * MDIM * 4);
    (void)ws_size; (void)in_sizes; (void)n_in; (void)out_size;

    gate_kernel   <<<S_TOK, 64,  0, stream>>>(x, wg, gates, topk, gk);
    me_kernel     <<<GEXP,  256, 0, stream>>>(gates, me);
    loc_kernel    <<<GEXP,  32,  0, stream>>>(topk, loc, ce);

    const long nzero = (long)GEXP * CAPAC * MDIM * 2 / 16;   // uint4 count
    zero_kernel   <<<2048, 256, 0, stream>>>((uint4*)disp, nzero);
    dispatch_kernel<<<S_TOK * KTOP, 128, 0, stream>>>(x, topk, loc, disp);
    convert_we_kernel<<<4096, 256, 0, stream>>>(we, web, (long)GEXP * MDIM * MDIM / 8);

    expert_gemm_kernel<<<GEXP * (CAPAC / BM) * (MDIM / BN), 256, 0, stream>>>(disp, web, eo);

    combine_kernel<<<S_TOK, 128, 0, stream>>>(eo, topk, loc, gk, y);
    loss_kernel   <<<1, GEXP, 0, stream>>>(me, ce, lloss);
}
